// SAGE_Full_46918222742091
// MI455X (gfx1250) — matrix operand order repaired and checked by reading, not compiled, not run on hardware
//
#include <hip/hip_runtime.h>

typedef __attribute__((ext_vector_type(16))) __bf16 v16bf;
typedef __attribute__((ext_vector_type(8)))  __bf16 v8bf;
typedef __attribute__((ext_vector_type(8)))  float  v8f;
typedef __attribute__((ext_vector_type(4)))  int    i4v;

#define N_NODES 50000
#define N_EDGES 800000
#define D 128
#define LDA 136   // padded LDS row stride (bf16): 272B rows; 136 = 17*8 -> v8bf-indexable

typedef __attribute__((address_space(1))) i4v g_i4;   // global int4
typedef __attribute__((address_space(3))) i4v l_i4;   // LDS int4

#if defined(__gfx1250__) && __has_builtin(__builtin_amdgcn_global_load_async_to_lds_b128)
#define HAVE_ASYNC_LDS 1
#else
#define HAVE_ASYNC_LDS 0
#endif

__device__ __forceinline__ void async_copy16(const void* g, void* l) {
#if HAVE_ASYNC_LDS
    __builtin_amdgcn_global_load_async_to_lds_b128((g_i4*)g, (l_i4*)l, 0, 0);
#else
    *(uint4*)l = *(const uint4*)g;
#endif
}

__device__ __forceinline__ void async_wait0() {
#if HAVE_ASYNC_LDS
#if __has_builtin(__builtin_amdgcn_s_wait_asynccnt)
    __builtin_amdgcn_s_wait_asynccnt(0);
#else
    asm volatile("s_wait_asynccnt 0x0" ::: "memory");
#endif
#endif
}

__device__ __forceinline__ v16bf cat8(v8bf lo, v8bf hi) {
    return __builtin_shufflevector(lo, hi, 0, 1, 2, 3, 4, 5, 6, 7,
                                           8, 9, 10, 11, 12, 13, 14, 15);
}

// ---------------- zero fill (sum + deg re-zeroed every call) -----------------------
__global__ void zero_f32(float* __restrict__ p, int n) {
    int i = blockIdx.x * blockDim.x + threadIdx.x;
    if (i < n) p[i] = 0.0f;
}

// ------------- fp32 -> bf16 weight conversion, TRANSPOSED to [N][K] ----------------
__global__ void convert_w(const float* __restrict__ Wself, const float* __restrict__ Wneigh,
                          __bf16* __restrict__ Wst, __bf16* __restrict__ Wnt) {
    int i = blockIdx.x * blockDim.x + threadIdx.x;   // 0..16383
    int k = i >> 7;          // source row   (K)
    int n = i & 127;         // source col   (N)
    Wst[n * D + k] = (__bf16)Wself[i];
    Wnt[n * D + k] = (__bf16)Wneigh[i];
}

// ---------------- edge-parallel mean-aggregation (scatter-add, L2-resident) --------
__global__ void aggregate(const float* __restrict__ h, const int* __restrict__ src,
                          const int* __restrict__ dst, float* __restrict__ sum,
                          float* __restrict__ deg) {
    int lane = threadIdx.x & 31;
    int wave = threadIdx.x >> 5;
    int e = blockIdx.x * 8 + wave;
    if (e >= N_EDGES) return;

    if (lane == 0) {
        int pe = e + 4096;
        if (pe < N_EDGES) {
            __builtin_prefetch(&src[pe], 0, 1);   // -> global_prefetch_b8
            __builtin_prefetch(&dst[pe], 0, 1);
        }
    }

    int s = src[e];
    int d = dst[e];
    const float4* hp = (const float4*)(h + (size_t)s * D);
    float4 v = hp[lane];
    float* sp = sum + (size_t)d * D + lane * 4;
    atomicAdd(sp + 0, v.x);
    atomicAdd(sp + 1, v.y);
    atomicAdd(sp + 2, v.z);
    atomicAdd(sp + 3, v.w);
    if (lane == 0) atomicAdd(deg + d, 1.0f);
}

// ---------------- fused normalize + dual-GEMM + bias + relu via WMMA bf16 ----------
// block = 256 threads = 8 waves; computes 16 rows x 128 cols.
// Weights (transposed bf16) staged to LDS via async b128 loads (ASYNCcnt path);
// every WMMA operand is exactly 2x ds_load_b128 (explicit v8bf vector loads).
__global__ void sage_gemm(const float* __restrict__ h, const float* __restrict__ sum,
                          const float* __restrict__ deg,
                          const __bf16* __restrict__ Wst, const __bf16* __restrict__ Wnt,
                          const float* __restrict__ bias, float* __restrict__ out,
                          int relu) {
    __shared__ __align__(16) __bf16 As[16 * LDA];     // self features
    __shared__ __align__(16) __bf16 An[16 * LDA];     // mean neighbor features
    __shared__ __align__(16) __bf16 Ls[128 * LDA];    // W_self^T  [N][K], padded rows
    __shared__ __align__(16) __bf16 Ln[128 * LDA];    // W_neigh^T [N][K], padded rows

    const int tid  = threadIdx.x;
    const int row0 = blockIdx.x * 16;

    // ---- async-stage both weight matrices: 2048 x 16B transfers each ----
    #pragma unroll
    for (int i = 0; i < 8; ++i) {
        int idx = tid + i * 256;            // 0..2047
        int r = idx >> 4, t = idx & 15;
        const char* gs = (const char*)Wst + idx * 16;
        const char* gn = (const char*)Wnt + idx * 16;
        char* ls = (char*)Ls + r * (LDA * 2) + t * 16;
        char* ln = (char*)Ln + r * (LDA * 2) + t * 16;
        async_copy16(gs, ls);
        async_copy16(gn, ln);
    }

    // ---- stage + convert A tiles (overlaps with async weight DMA) ----
    {
        int r  = tid >> 4;
        int c0 = (tid & 15) * 8;
        size_t base = (size_t)(row0 + r) * D + c0;
        float inv = 1.0f / fmaxf(deg[row0 + r], 1.0f);
        #pragma unroll
        for (int i = 0; i < 8; ++i) {
            As[r * LDA + c0 + i] = (__bf16)h[base + i];
            An[r * LDA + c0 + i] = (__bf16)(sum[base + i] * inv);
        }
    }
    async_wait0();
    __syncthreads();

    const int lane  = tid & 31;
    const int wave  = tid >> 5;
    const int cbase = wave * 16;
    const int m     = lane & 15;
    const int half  = lane >> 4;

    float bv = bias[cbase + m];
    v8f acc;
    #pragma unroll
    for (int i = 0; i < 8; ++i) acc[i] = bv;

    // per-lane row base pointers (all 16B aligned: row stride = 17 v8bf chunks)
    const v8bf* rowAs = (const v8bf*)(As + m * LDA);
    const v8bf* rowAn = (const v8bf*)(An + m * LDA);
    const v8bf* rowBs = (const v8bf*)(Ls + (cbase + m) * LDA);
    const v8bf* rowBn = (const v8bf*)(Ln + (cbase + m) * LDA);

    #pragma unroll
    for (int kt = 0; kt < 4; ++kt) {
        // A 16x32 bf16 layout: chunk0 = K[kt*32+8h .. +7], chunk1 = K[kt*32+16+8h .. +7]
        v16bf a_s = cat8(rowAs[kt * 4 + half], rowAs[kt * 4 + half + 2]);
        v16bf a_n = cat8(rowAn[kt * 4 + half], rowAn[kt * 4 + half + 2]);
        // B 32x16 bf16 layout: chunk0 = K[kt*32+8h .. +7], chunk1 = K[kt*32+16+8h .. +7]
        v16bf b_s = cat8(rowBs[kt * 4 + half], rowBs[kt * 4 + half + 2]);
        v16bf b_n = cat8(rowBn[kt * 4 + half], rowBn[kt * 4 + half + 2]);

        acc = __builtin_amdgcn_wmma_f32_16x16x32_bf16(false, a_s, false, b_s,
                                                      (short)0, acc, false, false);
        acc = __builtin_amdgcn_wmma_f32_16x16x32_bf16(false, a_n, false, b_n,
                                                      (short)0, acc, false, false);
    }

    #pragma unroll
    for (int r2 = 0; r2 < 8; ++r2) {
        int orow = row0 + r2 + half * 8;
        float v = acc[r2];
        if (relu) v = fmaxf(v, 0.0f);
        out[(size_t)orow * D + cbase + m] = v;
    }
}

extern "C" void kernel_launch(void* const* d_in, const int* in_sizes, int n_in,
                              void* d_out, int out_size, void* d_ws, size_t ws_size,
                              hipStream_t stream) {
    const float* feat = (const float*)d_in[0];
    const int*   esrc = (const int*)d_in[1];
    const int*   edst = (const int*)d_in[2];
    const float* Wself[3]  = {(const float*)d_in[3], (const float*)d_in[6], (const float*)d_in[9]};
    const float* Wneigh[3] = {(const float*)d_in[4], (const float*)d_in[7], (const float*)d_in[10]};
    const float* bias[3]   = {(const float*)d_in[5], (const float*)d_in[8], (const float*)d_in[11]};

    char* ws = (char*)d_ws;
    float*  h1   = (float*)ws;  ws += (size_t)N_NODES * D * sizeof(float);
    float*  h2   = (float*)ws;  ws += (size_t)N_NODES * D * sizeof(float);
    float*  sumb = (float*)ws;  ws += (size_t)N_NODES * D * sizeof(float);
    float*  degb = (float*)ws;  ws += (size_t)N_NODES * sizeof(float);   // adjacent to sumb
    __bf16* Wst  = (__bf16*)ws; ws += (size_t)D * D * sizeof(__bf16);
    __bf16* Wnt  = (__bf16*)ws; ws += (size_t)D * D * sizeof(__bf16);

    const float* hin[3]  = {feat, h1, h2};
    float*       hout[3] = {h1, h2, (float*)d_out};

    for (int l = 0; l < 3; ++l) {
        convert_w<<<64, 256, 0, stream>>>(Wself[l], Wneigh[l], Wst, Wnt);
        int nz = N_NODES * D + N_NODES;               // sum + deg (contiguous)
        zero_f32<<<(nz + 255) / 256, 256, 0, stream>>>(sumb, nz);
        aggregate<<<N_EDGES / 8, 256, 0, stream>>>(hin[l], esrc, edst, sumb, degb);
        sage_gemm<<<N_NODES / 16, 256, 0, stream>>>(hin[l], sumb, degb, Wst, Wnt,
                                                    bias[l], hout[l], (l < 2) ? 1 : 0);
    }
}
